// MultiHeadAttentionUnit_58969900974766
// MI455X (gfx1250) — matrix operand order repaired and checked by reading, not compiled, not run on hardware
//
#include <hip/hip_runtime.h>
#include <stdint.h>

#define Bn 4
#define Sn 2048
#define En 1024
#define Hn 16
#define Dn 64

typedef __attribute__((ext_vector_type(16))) __bf16 bf16x16;
typedef __attribute__((ext_vector_type(8)))  float        f32x8;
typedef __attribute__((ext_vector_type(8)))  unsigned int u32x8;

union F8 { f32x8 v; float f[8]; };

__device__ __forceinline__ unsigned short f2bf(float x) {
  unsigned int u = __builtin_bit_cast(unsigned int, x);
  unsigned int r = (u + 0x7FFFu + ((u >> 16) & 1u)) >> 16;
  return (unsigned short)r;
}

// Low 32 bits of a generic pointer to LDS == LDS byte offset (ISA 10.2 aperture map).
__device__ __forceinline__ unsigned lds_off(const void* p) {
  return (unsigned)(uintptr_t)p;
}

// Per-lane async copy of 16B: global (SGPR base + VGPR offset) -> LDS. ASYNCcnt.
__device__ __forceinline__ void async_copy_b128(unsigned lds_addr, unsigned goff,
                                                const void* sbase) {
  asm volatile("global_load_async_to_lds_b128 %0, %1, %2"
               :: "v"(lds_addr), "v"(goff), "s"(sbase)
               : "memory");
}
__device__ __forceinline__ void wait_async0() {
  asm volatile("s_wait_asynccnt 0x0" ::: "memory");
}

// A fragment (16xK=32 bf16) from row-major LDS tile [16][ldk], 4B-aligned reads.
// ISA 7.12.2: lane m = lane&15, group g = lane>>4; VGPR i holds K = 2(i&3)+16(i>>2)+8g, +1
__device__ __forceinline__ u32x8 load_a_frag(const unsigned short* base, int ldk) {
  int lane = threadIdx.x & 31;
  int g = lane >> 4, m = lane & 15;
  u32x8 r;
#pragma unroll
  for (int i = 0; i < 8; ++i) {
    int k = ((i & 3) << 1) + ((i >> 2) << 4) + (g << 3);
    r[i] = *(const unsigned int*)(base + m * ldk + k);
  }
  return r;
}

// B fragment (K=32 x 16 bf16) from LDS tile stored TRANSPOSED as [N=16][ldk(K)].
// VGPR j holds K = 2(j&3)+16(j>>2)+8g, +1 (N=lane&15, g=lane>>4)
__device__ __forceinline__ u32x8 load_bt_frag(const unsigned short* base, int ldk) {
  int lane = threadIdx.x & 31;
  int g = lane >> 4, n = lane & 15;
  u32x8 r;
#pragma unroll
  for (int j = 0; j < 8; ++j) {
    int k = ((j & 3) << 1) + ((j >> 2) << 4) + (g << 3);
    r[j] = *(const unsigned int*)(base + n * ldk + k);
  }
  return r;
}

__device__ __forceinline__ f32x8 wmma_bf(u32x8 a, u32x8 b, f32x8 c) {
  return __builtin_amdgcn_wmma_f32_16x16x32_bf16(
      false, __builtin_bit_cast(bf16x16, a),
      false, __builtin_bit_cast(bf16x16, b),
      (short)0, c, false, false);
}

// ---------------------------------------------------------------------------
// Kernel A: per-head 2-layer MLP  out = (relu(x@W1 + b1) @ W2 + b2) * scale
// grid (S/128, H, B), 256 threads. Output bf16 [B,H,S,D].
// (f32->bf16 conversion happens while staging, so staging stays on the VGPR path.)
// ---------------------------------------------------------------------------
__global__ __launch_bounds__(256) void qkv_mlp_kernel(
    const float* __restrict__ x,  const float* __restrict__ W1,
    const float* __restrict__ b1, const float* __restrict__ W2,
    const float* __restrict__ b2, unsigned short* __restrict__ outbf,
    float out_scale)
{
  __shared__ unsigned short s_x[128][32];   // A tile (bf16)
  __shared__ unsigned short s_w[64][32];    // B^T tile (bf16)
  __shared__ unsigned short s_h[128][64];   // hidden (bf16), A of layer 2

  const int tid  = threadIdx.x;
  const int wave = tid >> 5, lane = tid & 31;
  const int g = lane >> 4, ln = lane & 15;
  const int s0 = blockIdx.x * 128;
  const int h  = blockIdx.y;
  const int b  = blockIdx.z;
  const int m0 = wave * 16;

  F8 acc[4];
#pragma unroll
  for (int nt = 0; nt < 4; ++nt)
#pragma unroll
    for (int r = 0; r < 8; ++r) acc[nt].f[r] = 0.0f;

  // ---- layer 1: hidden = relu(x[128,1024] @ W1[1024,64] + b1) ----
  for (int k0 = 0; k0 < En; k0 += 32) {
    for (int idx = tid; idx < 128 * 32; idx += 256) {
      int row = idx >> 5, col = idx & 31;
      s_x[row][col] = f2bf(x[(size_t)(b * Sn + s0 + row) * En + k0 + col]);
    }
    for (int idx = tid; idx < 64 * 32; idx += 256) {
      int n = idx & 63, kk = idx >> 6;                       // n fastest: coalesced
      s_w[n][kk] = f2bf(W1[((size_t)h * En + k0 + kk) * Dn + n]);
    }
    __syncthreads();
    u32x8 a = load_a_frag(&s_x[m0][0], 32);
#pragma unroll
    for (int nt = 0; nt < 4; ++nt) {
      u32x8 bb = load_bt_frag(&s_w[nt * 16][0], 32);
      acc[nt].v = wmma_bf(a, bb, acc[nt].v);
    }
    __syncthreads();
  }
  // bias + relu -> s_h (bf16)
#pragma unroll
  for (int nt = 0; nt < 4; ++nt) {
    float bv = b1[h * Dn + nt * 16 + ln];
#pragma unroll
    for (int r = 0; r < 8; ++r) {
      float v = acc[nt].f[r] + bv;
      v = v > 0.0f ? v : 0.0f;
      s_h[m0 + g * 8 + r][nt * 16 + ln] = f2bf(v);
    }
  }
  __syncthreads();

  // ---- layer 2: out = hidden[128,64] @ W2[64,64] + b2 ----
  F8 acc2[4];
#pragma unroll
  for (int nt = 0; nt < 4; ++nt)
#pragma unroll
    for (int r = 0; r < 8; ++r) acc2[nt].f[r] = 0.0f;

  for (int k0 = 0; k0 < Dn; k0 += 32) {
    for (int idx = tid; idx < 64 * 32; idx += 256) {
      int n = idx & 63, kk = idx >> 6;
      s_w[n][kk] = f2bf(W2[((size_t)h * Dn + k0 + kk) * Dn + n]);
    }
    __syncthreads();
    u32x8 a = load_a_frag(&s_h[m0][k0], 64);
#pragma unroll
    for (int nt = 0; nt < 4; ++nt) {
      u32x8 bb = load_bt_frag(&s_w[nt * 16][0], 32);
      acc2[nt].v = wmma_bf(a, bb, acc2[nt].v);
    }
    __syncthreads();
  }
#pragma unroll
  for (int nt = 0; nt < 4; ++nt) {
    float bv = b2[h * Dn + nt * 16 + ln];
#pragma unroll
    for (int r = 0; r < 8; ++r) {
      int row = s0 + m0 + g * 8 + r;
      float v = (acc2[nt].f[r] + bv) * out_scale;
      outbf[((size_t)(b * Hn + h) * Sn + row) * Dn + nt * 16 + ln] = f2bf(v);
    }
  }
}

// ---------------------------------------------------------------------------
// Kernel B: causal flash attention per (b,h). Q is pre-scaled by 1/sqrt(D).
// grid (S/128, H, B), 256 threads. concat output bf16 [B,S,H*D].
// Q tile + K chunks staged via GLOBAL_LOAD_ASYNC_TO_LDS_B128 (ASYNCcnt).
// ---------------------------------------------------------------------------
__global__ __launch_bounds__(256) void attention_kernel(
    const unsigned short* __restrict__ Q,
    const unsigned short* __restrict__ K,
    const unsigned short* __restrict__ V,
    unsigned short* __restrict__ concat)
{
  __shared__ unsigned short s_q[128][64];   // Q tile (A layout)
  __shared__ unsigned short s_k[64][64];    // K chunk row-major == B^T for Q@K^T
  __shared__ unsigned short s_vt[64][64];   // V^T chunk == B^T for P@V
  __shared__ unsigned short s_p[128][64];   // P (per-wave exclusive 16-row slab)

  const int tid  = threadIdx.x;
  const int wave = tid >> 5, lane = tid & 31;
  const int g = lane >> 4, ln = lane & 15;
  const int s0 = blockIdx.x * 128, h = blockIdx.y, b = blockIdx.z;
  const int m0 = wave * 16;
  const size_t headbase = (size_t)(b * Hn + h) * Sn * Dn;

  // async-stage Q tile: 128*64*2 = 16384 B = 1024 x b128, 4 per lane
  {
    const void* qg = (const void*)(Q + headbase + (size_t)s0 * Dn);
    unsigned ldsQ = lds_off(&s_q[0][0]);
#pragma unroll
    for (int t = 0; t < 4; ++t) {
      unsigned off = (unsigned)(tid + t * 256) * 16u;
      async_copy_b128(ldsQ + off, off, qg);
    }
  }

  float mi[8], li[8];
  F8 o[4];
#pragma unroll
  for (int r = 0; r < 8; ++r) { mi[r] = -3.0e38f; li[r] = 0.0f; }
#pragma unroll
  for (int nt = 0; nt < 4; ++nt)
#pragma unroll
    for (int r = 0; r < 8; ++r) o[nt].f[r] = 0.0f;

  const int nchunk = (s0 >> 6) + 2;                 // keys 0 .. s0+127
  for (int c = 0; c < nchunk; ++c) {
    const int j0 = c * 64;
    __syncthreads();                                // protect s_k/s_vt re-stage
    // async-stage K chunk: 64*64*2 = 8192 B = 512 x b128, 2 per lane
    {
      const void* kg = (const void*)(K + headbase + (size_t)j0 * Dn);
      unsigned ldsK = lds_off(&s_k[0][0]);
#pragma unroll
      for (int t = 0; t < 2; ++t) {
        unsigned off = (unsigned)(tid + t * 256) * 16u;
        async_copy_b128(ldsK + off, off, kg);
      }
    }
    // manual transposed stage of V chunk
    for (int idx = tid; idx < 64 * 64; idx += 256) {
      int kk = idx >> 6, d = idx & 63;
      s_vt[d][kk] = V[headbase + (size_t)(j0 + kk) * Dn + d];
    }
    wait_async0();                                  // Q (first iter) + K complete
    __syncthreads();

    if (j0 <= s0 + m0 + 15) {                       // wave-uniform causal skip
      F8 sc[4];
#pragma unroll
      for (int nt = 0; nt < 4; ++nt)
#pragma unroll
        for (int r = 0; r < 8; ++r) sc[nt].f[r] = 0.0f;
#pragma unroll
      for (int d0 = 0; d0 < 64; d0 += 32) {
        u32x8 a = load_a_frag(&s_q[m0][d0], 64);
#pragma unroll
        for (int nt = 0; nt < 4; ++nt) {
          u32x8 bb = load_bt_frag(&s_k[nt * 16][d0], 64);
          sc[nt].v = wmma_bf(a, bb, sc[nt].v);
        }
      }
      // online softmax (rows live across 16-lane groups in C layout)
#pragma unroll
      for (int r = 0; r < 8; ++r) {
        int qg2 = s0 + m0 + g * 8 + r;
        float sv[4];
        float rowmax = -3.0e38f;
#pragma unroll
        for (int nt = 0; nt < 4; ++nt) {
          int kg2 = j0 + nt * 16 + ln;
          float s = sc[nt].f[r];
          s = (kg2 <= qg2) ? s : -3.0e38f;
          sv[nt] = s;
          rowmax = fmaxf(rowmax, s);
        }
#pragma unroll
        for (int off = 8; off >= 1; off >>= 1)
          rowmax = fmaxf(rowmax, __shfl_xor(rowmax, off, 32));
        float mnew  = fmaxf(mi[r], rowmax);
        float scale = __expf(mi[r] - mnew);
        float psum  = 0.0f;
#pragma unroll
        for (int nt = 0; nt < 4; ++nt) {
          float p = __expf(sv[nt] - mnew);
          psum += p;
          s_p[m0 + g * 8 + r][nt * 16 + ln] = f2bf(p);
        }
#pragma unroll
        for (int off = 8; off >= 1; off >>= 1)
          psum += __shfl_xor(psum, off, 32);
        li[r] = li[r] * scale + psum;
        mi[r] = mnew;
#pragma unroll
        for (int nt = 0; nt < 4; ++nt) o[nt].f[r] *= scale;
      }
      // O += P[16,64] @ V[64,64]  (P read back from own LDS slab: in-wave DS order)
#pragma unroll
      for (int kc = 0; kc < 64; kc += 32) {
        u32x8 a = load_a_frag(&s_p[m0][kc], 64);
#pragma unroll
        for (int nt = 0; nt < 4; ++nt) {
          u32x8 bb = load_bt_frag(&s_vt[nt * 16][kc], 64);
          o[nt].v = wmma_bf(a, bb, o[nt].v);
        }
      }
    }
  }
  // normalize + store into concat layout [B,S,H*D]
#pragma unroll
  for (int nt = 0; nt < 4; ++nt) {
#pragma unroll
    for (int r = 0; r < 8; ++r) {
      int row = s0 + m0 + g * 8 + r;
      float v = o[nt].f[r] / li[r];
      concat[((size_t)(b * Sn + row)) * (Hn * Dn) + h * Dn + nt * 16 + ln] = f2bf(v);
    }
  }
}

// ---------------------------------------------------------------------------
// Kernel C: out = concat[8192,1024] @ oW[1024,1024] + ob  (fp32 output)
// grid (8192/128, 1024/64), 256 threads. A tile staged async (bf16 pass-through).
// ---------------------------------------------------------------------------
__global__ __launch_bounds__(256) void outproj_kernel(
    const unsigned short* __restrict__ concat,
    const float* __restrict__ oW, const float* __restrict__ ob,
    float* __restrict__ out)
{
  __shared__ unsigned short s_a[128][32];
  __shared__ unsigned short s_w[64][32];
  const int tid  = threadIdx.x;
  const int wave = tid >> 5, lane = tid & 31;
  const int g = lane >> 4, ln = lane & 15;
  const int r0 = blockIdx.x * 128, n0 = blockIdx.y * 64;
  const int m0 = wave * 16;
  const int N = Hn * Dn;

  F8 acc[4];
#pragma unroll
  for (int nt = 0; nt < 4; ++nt)
#pragma unroll
    for (int r = 0; r < 8; ++r) acc[nt].f[r] = 0.0f;

  const unsigned ldsA = lds_off(&s_a[0][0]);
  for (int k0 = 0; k0 < N; k0 += 32) {
    // async-stage A tile: 128 rows x 64B; row segment = 16B, 512 x b128, 2/lane
    {
      const void* ag = (const void*)(concat + (size_t)r0 * N + k0);
#pragma unroll
      for (int t = 0; t < 2; ++t) {
        int tt = tid + t * 256;                    // 0..511
        int row = tt >> 2, seg = tt & 3;
        unsigned goff = (unsigned)(row * N + seg * 8) * 2u;   // bytes in global
        unsigned loff = (unsigned)(row * 32 + seg * 8) * 2u;  // bytes in LDS
        async_copy_b128(ldsA + loff, goff, ag);
      }
    }
    for (int idx = tid; idx < 64 * 32; idx += 256) {
      int n = idx & 63, kk = idx >> 6;
      s_w[n][kk] = f2bf(oW[(size_t)(k0 + kk) * N + n0 + n]);
    }
    wait_async0();
    __syncthreads();
    u32x8 a = load_a_frag(&s_a[m0][0], 32);
#pragma unroll
    for (int nt = 0; nt < 4; ++nt) {
      u32x8 bb = load_bt_frag(&s_w[nt * 16][0], 32);
      acc[nt].v = wmma_bf(a, bb, acc[nt].v);
    }
    __syncthreads();
  }
#pragma unroll
  for (int nt = 0; nt < 4; ++nt) {
    float bv = ob[n0 + nt * 16 + ln];
#pragma unroll
    for (int r = 0; r < 8; ++r) {
      out[(size_t)(r0 + m0 + g * 8 + r) * N + n0 + nt * 16 + ln] = acc[nt].f[r] + bv;
    }
  }
}

// ---------------------------------------------------------------------------
extern "C" void kernel_launch(void* const* d_in, const int* in_sizes, int n_in,
                              void* d_out, int out_size, void* d_ws, size_t ws_size,
                              hipStream_t stream) {
  const float* x   = (const float*)d_in[0];
  const float* qW1 = (const float*)d_in[1];
  const float* qb1 = (const float*)d_in[2];
  const float* qW2 = (const float*)d_in[3];
  const float* qb2 = (const float*)d_in[4];
  const float* kW1 = (const float*)d_in[5];
  const float* kb1 = (const float*)d_in[6];
  const float* kW2 = (const float*)d_in[7];
  const float* kb2 = (const float*)d_in[8];
  const float* vW1 = (const float*)d_in[9];
  const float* vb1 = (const float*)d_in[10];
  const float* vW2 = (const float*)d_in[11];
  const float* vb2 = (const float*)d_in[12];
  const float* oW  = (const float*)d_in[13];
  const float* ob  = (const float*)d_in[14];

  unsigned short* ws = (unsigned short*)d_ws;
  const size_t qkvElems = (size_t)Bn * Hn * Sn * Dn;  // 8,388,608
  unsigned short* Qb = ws;
  unsigned short* Kb = ws + qkvElems;
  unsigned short* Vb = ws + 2 * qkvElems;
  unsigned short* Cc = ws + 3 * qkvElems;             // [B,S,H*D] bf16

  dim3 blk(256);
  dim3 gA(Sn / 128, Hn, Bn);
  // fold 1/sqrt(D)=0.125 into Q
  qkv_mlp_kernel<<<gA, blk, 0, stream>>>(x, qW1, qb1, qW2, qb2, Qb, 0.125f);
  qkv_mlp_kernel<<<gA, blk, 0, stream>>>(x, kW1, kb1, kW2, kb2, Kb, 1.0f);
  qkv_mlp_kernel<<<gA, blk, 0, stream>>>(x, vW1, vb1, vW2, vb2, Vb, 1.0f);
  attention_kernel<<<dim3(Sn / 128, Hn, Bn), blk, 0, stream>>>(Qb, Kb, Vb, Cc);
  outproj_kernel<<<dim3((Bn * Sn) / 128, (Hn * Dn) / 64), blk, 0, stream>>>(
      Cc, oW, ob, (float*)d_out);
}
